// CA_global_40415642255795
// MI455X (gfx1250) — matrix operand order repaired and checked by reading, not compiled, not run on hardware
//
#include <hip/hip_runtime.h>

typedef _Float16 h16;
typedef __attribute__((ext_vector_type(8)))  _Float16 v8h;
typedef __attribute__((ext_vector_type(16))) _Float16 v16h;
typedef __attribute__((ext_vector_type(2)))  float    v2f;
typedef __attribute__((ext_vector_type(8)))  float    v8f;

#define NH      4
#define XPITCH  68   // floats per (ch,row) line of s_xy (16B aligned, bank-skewed)
#define APITCH  72   // halves per pos-line of activations
#define QPITCH  65   // floats per row of qkv (row*4 stride != 0 mod 64 banks)

// ---------------------------------------------------------------------------
// Cross-lane xor within 16-lane rows via DPP16 ROW_XMASK (VALU, no LDS, no
// counters).
// ---------------------------------------------------------------------------
template <int MASK>
__device__ inline float dpp_xor16(float x) {
  int r = __builtin_amdgcn_update_dpp(0, __float_as_int(x),
                                      0x160 | MASK /*ROW_XMASK*/, 0xF, 0xF, true);
  return __int_as_float(r);
}
__device__ inline float red_max16(float x) {
  x = fmaxf(x, dpp_xor16<1>(x));
  x = fmaxf(x, dpp_xor16<2>(x));
  x = fmaxf(x, dpp_xor16<4>(x));
  x = fmaxf(x, dpp_xor16<8>(x));
  return x;
}
__device__ inline float red_sum16(float x) {
  x += dpp_xor16<1>(x);
  x += dpp_xor16<2>(x);
  x += dpp_xor16<4>(x);
  x += dpp_xor16<8>(x);
  return x;
}

// ---------------------------------------------------------------------------
// QKV projection: out[o][n] = sum_ic W[o][ic] * act[ic][n]  (f16 WMMA, f32 acc)
// IC = 64 (low/high, K-tiles 2) or 128 (mid, K-tiles 4). N = 64 positions.
// A-frag (f16 16x32): lane m = lane%16, VGPR v holds K = 2v + (v>=4?8:0) + (lane>=16?8:0)
//   -> per lane two contiguous float-8 runs of the weight row.
// B-frag (f16 32x16): lane n = lane%16, VGPR v holds K = 2v + (v>=4?8:0) + (lane>=16?8:0)
//   -> two 8-half runs of act[pos][*] (same K order as A).
// ---------------------------------------------------------------------------
template <int IC>
__device__ inline void gemm_branch(const float* __restrict__ W,
                                   const h16 (*act0)[APITCH],
                                   const h16 (*act1)[APITCH],
                                   float (*qkv)[QPITCH],
                                   int wid, int lmod, int lhi) {
  constexpr int KT = IC / 32;        // k-tiles
  constexpr int MT = 3 * IC / 16;    // m-tiles (3*IC output channels)
  for (int mt = wid; mt < MT; mt += 8) {
    v16h af[KT];
#pragma unroll
    for (int kt = 0; kt < KT; ++kt) {
      const float* wr = W + (mt * 16 + lmod) * IC + kt * 32 + lhi * 8;
      float4 w0 = *(const float4*)(wr);
      float4 w1 = *(const float4*)(wr + 4);
      float4 w2 = *(const float4*)(wr + 16);
      float4 w3 = *(const float4*)(wr + 20);
      v16h a;
      a[0]  = (h16)w0.x; a[1]  = (h16)w0.y; a[2]  = (h16)w0.z; a[3]  = (h16)w0.w;
      a[4]  = (h16)w1.x; a[5]  = (h16)w1.y; a[6]  = (h16)w1.z; a[7]  = (h16)w1.w;
      a[8]  = (h16)w2.x; a[9]  = (h16)w2.y; a[10] = (h16)w2.z; a[11] = (h16)w2.w;
      a[12] = (h16)w3.x; a[13] = (h16)w3.y; a[14] = (h16)w3.z; a[15] = (h16)w3.w;
      af[kt] = a;
    }
#pragma unroll
    for (int nt = 0; nt < 4; ++nt) {
      v8f acc = {};
      const int pos = nt * 16 + lmod;
#pragma unroll
      for (int kt = 0; kt < KT; ++kt) {
        const h16(*A)[APITCH] = (kt < 2) ? act0 : act1;
        const h16* bp = &A[pos][(kt & 1) * 32 + lhi * 8];
        v8h blo = *(const v8h*)bp;
        v8h bhi = *(const v8h*)(bp + 16);
        v16h bf;
#pragma unroll
        for (int i = 0; i < 8; ++i) { bf[i] = blo[i]; bf[8 + i] = bhi[i]; }
        acc = __builtin_amdgcn_wmma_f32_16x16x32_f16(false, af[kt], false, bf,
                                                     (short)0, acc, false, false);
      }
#pragma unroll
      for (int v = 0; v < 8; ++v)
        qkv[mt * 16 + v + 8 * lhi][nt * 16 + lmod] = acc[v];
    }
  }
}

// ---------------------------------------------------------------------------
// Channel attention per 2x2 window + fused inverse-DWT scatter.
// CC = per-head channels (16 low/high, 32 mid). Contraction dim s=4 ->
// v_wmma_f32_16x16x4_f32 per (window, head) tile.
// A (f32 16x4): lane m = lane%16, VGPR v -> K(=s) = v + 2*(lane/16)
// B (f32 4x16): lane n = lane%16, VGPR v -> K(=s) = v + 2*(lane/16)
// D (f32 16x16): M = v + 8*(lane/16) (=ci), N = lane%16 (=cd)
// ---------------------------------------------------------------------------
template <int CC>
__device__ inline void attn_branch(const float (*qkv)[QPITCH],
                                   float (*yacc)[4][XPITCH], float scale,
                                   int br, int wid, int lmod, int lhi) {
  constexpr int OC = CC * NH;    // 64 or 128
  constexpr int CT = CC / 16;    // 16x16 tiles per axis
  const int s0 = 2 * lhi;
  const int b0 = lmod & 1, b1 = (lmod >> 1) & 1;
  for (int p = wid; p < 64; p += 8) {   // 16 windows * 4 heads, uniform trip count
    const int win = p >> 2, hd = p & 3, nb = win * 4;
    v2f aq[CT], bk[CT];
#pragma unroll
    for (int t = 0; t < CT; ++t) {
      const int ci = t * 16 + lmod;
      aq[t][0] = scale * qkv[ci * NH + hd][nb + s0];
      aq[t][1] = scale * qkv[ci * NH + hd][nb + s0 + 1];
      bk[t][0] = qkv[OC + ci * NH + hd][nb + s0];
      bk[t][1] = qkv[OC + ci * NH + hd][nb + s0 + 1];
    }
    v8f dl[CT][CT];
#pragma unroll
    for (int tq = 0; tq < CT; ++tq)
#pragma unroll
      for (int td = 0; td < CT; ++td) {
        v8f z = {};
        dl[tq][td] = __builtin_amdgcn_wmma_f32_16x16x4_f32(
            false, aq[tq], false, bk[td], (short)0, z, false, false);
      }
    // softmax over cd: per-lane combine across column tiles BEFORE the
    // butterfly (red(max(x0,x1)) == max of full row; red(e0+e1) == row sum)
#pragma unroll
    for (int tq = 0; tq < CT; ++tq)
#pragma unroll
      for (int v = 0; v < 8; ++v) {
        float x0 = dl[tq][0][v];
        float xm = x0, x1 = 0.f;
        if constexpr (CT == 2) {
          x1 = dl[tq][1][v];
          xm = fmaxf(x0, x1);
        }
        const float rmax = red_max16(xm);
        float e0 = __expf(x0 - rmax);
        float es = e0, e1 = 0.f;
        if constexpr (CT == 2) {
          e1 = __expf(x1 - rmax);
          es = e0 + e1;
        }
        const float inv = 1.0f / red_sum16(es);
        dl[tq][0][v] = e0 * inv;
        if constexpr (CT == 2) dl[tq][1][v] = e1 * inv;
      }
    // P*V: folding butterfly — 4 per-lane products collapse to one value per
    // lane in 4 xor steps; slot s = 2*bit0 + bit1 of the lane ends on lanes 0..3.
    float vv[CT][4];
#pragma unroll
    for (int t = 0; t < CT; ++t)
#pragma unroll
      for (int s = 0; s < 4; ++s)
        vv[t][s] = qkv[2 * OC + (t * 16 + lmod) * NH + hd][nb + s];
#pragma unroll
    for (int tq = 0; tq < CT; ++tq)
#pragma unroll
      for (int v = 0; v < 8; ++v) {
        float t0 = dl[tq][0][v] * vv[0][0];
        float t1 = dl[tq][0][v] * vv[0][1];
        float t2 = dl[tq][0][v] * vv[0][2];
        float t3 = dl[tq][0][v] * vv[0][3];
        if constexpr (CT == 2) {
          t0 += dl[tq][1][v] * vv[1][0];
          t1 += dl[tq][1][v] * vv[1][1];
          t2 += dl[tq][1][v] * vv[1][2];
          t3 += dl[tq][1][v] * vv[1][3];
        }
        // fold step 1 (xor 1): even lanes keep slots {0,1}, odd keep {2,3}
        const float keepA = b0 ? t2 : t0, sendA = b0 ? t0 : t2;
        const float keepB = b0 ? t3 : t1, sendB = b0 ? t1 : t3;
        const float u0 = keepA + dpp_xor16<1>(sendA);
        const float u1 = keepB + dpp_xor16<1>(sendB);
        // fold step 2 (xor 2): bit1=0 keeps first, bit1=1 keeps second
        const float keepC = b1 ? u1 : u0, sendC = b1 ? u0 : u1;
        float w = keepC + dpp_xor16<2>(sendC);
        // plain butterfly for remaining lanes
        w += dpp_xor16<4>(w);
        w += dpp_xor16<8>(w);
        if (lmod < 4) {  // lane lmod holds out[m][s], s = 2*bit0 + bit1
          const int s = (b0 << 1) | b1;
          const int ci = tq * 16 + v + 8 * lhi;
          const int cc = ci * NH + hd;
          const float u = 0.5f * w;
          const int sy = s >> 1, sx = s & 1;
          const int r0 = 2 * sy, c0 = 4 * win + 2 * sx;
          if (br == 0) {                       // LL: + + + +  (first writer: assign)
            yacc[cc][r0][c0]     = u; yacc[cc][r0][c0 + 1]     = u;
            yacc[cc][r0 + 1][c0] = u; yacc[cc][r0 + 1][c0 + 1] = u;
          } else if (br == 2) {                // HH: + - - +
            yacc[cc][r0][c0]     += u; yacc[cc][r0][c0 + 1]     -= u;
            yacc[cc][r0 + 1][c0] -= u; yacc[cc][r0 + 1][c0 + 1] += u;
          } else if (cc < 64) {                // LH: - - + +
            yacc[cc][r0][c0]     -= u; yacc[cc][r0][c0 + 1]     -= u;
            yacc[cc][r0 + 1][c0] += u; yacc[cc][r0 + 1][c0 + 1] += u;
          } else {                             // HL: - + - +
            const int ch = cc - 64;
            yacc[ch][r0][c0]     -= u; yacc[ch][r0][c0 + 1]     += u;
            yacc[ch][r0 + 1][c0] -= u; yacc[ch][r0 + 1][c0 + 1] += u;
          }
        }
      }
  }
}

// ---------------------------------------------------------------------------
// Fused DWT -> QKV(WMMA f16) -> window channel-attention(WMMA f32 K=4) -> IDWT
// Grid: (4 col-groups, 64 window-rows, 8 batch). Block: 256 = 8 waves.
// Each WG: 64ch x 4row x 64col tile of x / y; 16 windows.
// ---------------------------------------------------------------------------
__global__ __launch_bounds__(256) void wavelet_window_attn(
    const float* __restrict__ x, const float* __restrict__ w_lh,
    const float* __restrict__ w_m, float* __restrict__ out) {
  __shared__ __align__(16) float s_xy[64][4][XPITCH];    // x stage, then y accum
  __shared__ __align__(16) h16   s_act[4][64][APITCH];   // LL, LH, HL, HH (f16)
  __shared__ __align__(16) float s_qkv[384][QPITCH];     // conv output (per branch)

  const int tid = threadIdx.x;
  const int lane = tid & 31, wid = tid >> 5;
  const int lmod = lane & 15, lhi = lane >> 4;
  const int wgx = blockIdx.x, hrow = blockIdx.y, b = blockIdx.z;
  const size_t imgBase = (size_t)b * 64 * 65536;
  const int row0 = 4 * hrow, col0 = 64 * wgx;

  // warm L2 for the (shared, small) weight matrices
  for (int i = tid * 64; i < 384 * 128; i += 256 * 64) __builtin_prefetch(w_m + i, 0, 3);
  for (int i = tid * 64; i < 192 * 64; i += 256 * 64) __builtin_prefetch(w_lh + i, 0, 3);

  // Phase 1: stage x tile (64ch x 4 x 64) into LDS, float4 coalesced
  for (int idx = tid; idx < 64 * 4 * 16; idx += 256) {
    const int c4 = idx & 15, r = (idx >> 4) & 3, ch = idx >> 6;
    const float4 vdat = *(const float4*)(x + imgBase + (size_t)ch * 65536 +
                                         (size_t)(row0 + r) * 256 + col0 + c4 * 4);
    *(float4*)&s_xy[ch][r][c4 * 4] = vdat;
  }
  __syncthreads();

  // Phase 2: Haar DWT -> f16 activations [pos][ch], pos = win*4 + sy*2 + sx
  for (int idx = tid; idx < 64 * 64; idx += 256) {
    const int ch = idx & 63, pos = idx >> 6;
    const int w = pos >> 2, s = pos & 3, sy = s >> 1, sx = s & 1;
    const int r0 = 2 * sy, c0 = 4 * w + 2 * sx;
    const float a = s_xy[ch][r0][c0],     bb = s_xy[ch][r0][c0 + 1];
    const float c = s_xy[ch][r0 + 1][c0], d  = s_xy[ch][r0 + 1][c0 + 1];
    s_act[0][pos][ch] = (h16)((a + bb + c + d) * 0.5f);   // LL
    s_act[1][pos][ch] = (h16)((-a - bb + c + d) * 0.5f);  // LH
    s_act[2][pos][ch] = (h16)((-a + bb - c + d) * 0.5f);  // HL
    s_act[3][pos][ch] = (h16)((a - bb - c + d) * 0.5f);   // HH
  }
  __syncthreads();

  // LOW branch (c=16, scale = 16^-0.5)
  gemm_branch<64>(w_lh, s_act[0], s_act[0], s_qkv, wid, lmod, lhi);
  __syncthreads();
  attn_branch<16>(s_qkv, s_xy, 0.25f, 0, wid, lmod, lhi);
  __syncthreads();
  // MID branch (c=32, scale = 32^-0.5), input = [LH ; HL]
  gemm_branch<128>(w_m, s_act[1], s_act[2], s_qkv, wid, lmod, lhi);
  __syncthreads();
  attn_branch<32>(s_qkv, s_xy, 0.17677669529663689f, 1, wid, lmod, lhi);
  __syncthreads();
  // HIGH branch (c=16)
  gemm_branch<64>(w_lh, s_act[3], s_act[3], s_qkv, wid, lmod, lhi);
  __syncthreads();
  attn_branch<16>(s_qkv, s_xy, 0.25f, 2, wid, lmod, lhi);
  __syncthreads();

  // Phase 6: store reconstructed tile
  for (int idx = tid; idx < 64 * 4 * 16; idx += 256) {
    const int c4 = idx & 15, r = (idx >> 4) & 3, ch = idx >> 6;
    *(float4*)(out + imgBase + (size_t)ch * 65536 + (size_t)(row0 + r) * 256 +
               col0 + c4 * 4) = *(const float4*)&s_xy[ch][r][c4 * 4];
  }
}

extern "C" void kernel_launch(void* const* d_in, const int* /*in_sizes*/,
                              int /*n_in*/, void* d_out, int /*out_size*/,
                              void* /*d_ws*/, size_t /*ws_size*/,
                              hipStream_t stream) {
  const float* x    = (const float*)d_in[0];
  const float* w_lh = (const float*)d_in[1];
  const float* w_m  = (const float*)d_in[2];
  dim3 grid(4, 64, 8), block(256);
  hipLaunchKernelGGL(wavelet_window_attn, grid, block, 0, stream, x, w_lh, w_m,
                     (float*)d_out);
}
